// Fedformer_30958124270048
// MI455X (gfx1250) — matrix operand order repaired and checked by reading, not compiled, not run on hardware
//
#include <hip/hip_runtime.h>
#include <hip/hip_bf16.h>

// ---------------- model constants ----------------
#define Bsz   4
#define Lseq  4096
#define Lpred 2048
#define Llab  2048
#define Ldec  4096
#define NFt   21
#define Dmod  512
#define Hn    8
#define En    64
#define Mn    64
#define MAw   25
#define Dff   2048
#define BLEN  (Bsz * Lseq)         /* 16384 rows; == Bsz*Ldec */
#define RBHE  (Bsz * Hn * En)      /* 2048 fourier rows */

#define CEILDIV(a, b) (((a) + (b) - 1) / (b))

typedef __bf16 bf16_t;
typedef bf16_t v16bf __attribute__((ext_vector_type(16)));
typedef float  v8f   __attribute__((ext_vector_type(8)));
typedef unsigned int u32x4 __attribute__((ext_vector_type(4)));
typedef unsigned int u32x2 __attribute__((ext_vector_type(2)));

union FragU { v16bf v; u32x4 q[2]; };

__device__ __forceinline__ unsigned short f2bf(float f) {
  unsigned u = __float_as_uint(f);
  u += 0x7fffu + ((u >> 16) & 1u);          // round-to-nearest-even
  return (unsigned short)(u >> 16);
}
__device__ __forceinline__ float gelu_f(float x) {
  return 0.5f * x * (1.0f + erff(x * 0.7071067811865475f));
}

// ---------------- WMMA GEMM: C = act(A[M,K] * W[N,K]^T + bias + res) ----------------
// FULL-TILE ONLY: requires M % 128 == 0, N % 128 == 0, K % 32 == 0 (true for all
// large GEMMs in this model). 256 threads = 8 wave32 waves; block tile 128x128x32;
// wave owns 2 M-tiles x 4 N-tiles -> 8 v_wmma_f32_16x16x32_bf16 per K-step.
// Global loads are register-double-buffered so the 8 global_load_b128 of tile k+1
// overlap the WMMA stage of tile k.
#define BM 128
#define BN 128
#define BK 32

__global__ __launch_bounds__(256) void gemm_wmma_kernel(
    const float* __restrict__ A, int lda,
    const float* __restrict__ W,          // (Ndim, Kdim) row-major weights
    const float* __restrict__ bias,       // nullable
    const float* __restrict__ res,        // nullable, same layout as C
    float* __restrict__ C, int ldc,
    int Kdim, int act)
{
  __shared__ __align__(16) unsigned short As[BM][BK + 8];   // [m][k], bf16 bits, 80B rows
  __shared__ __align__(16) unsigned short Bs[BN][BK + 8];   // [n][k], bf16 bits
  const int tid  = threadIdx.x;
  const int lane = tid & 31;
  const int wave = tid >> 5;
  const int hh   = lane >> 4;       // lane half (0/1)
  const int l15  = lane & 15;
  const int m0 = blockIdx.y * BM;
  const int n0 = blockIdx.x * BN;
  const int wm = wave & 3;          // M pair: tiles 2*wm, 2*wm+1
  const int wn = wave >> 2;         // N quad: tiles wn*4 .. wn*4+3

  // this thread's fixed (row, 4-col) slice of each staged tile
  const int ldr = tid >> 1;                  // 0..127
  const int ldc4 = (tid & 1) * 16;           // 0 or 16 (two float4 per thread per row pair)
  // simpler mapping actually used below: idx-based (see loops)

  v8f acc[2][4];
  #pragma unroll
  for (int i2 = 0; i2 < 2; ++i2)
    #pragma unroll
    for (int j = 0; j < 4; ++j)
      #pragma unroll
      for (int r = 0; r < 8; ++r) acc[i2][j][r] = 0.0f;

  float4 ra[4], rb[4];
  // ---- prologue: load K-tile 0 into registers (8x global_load_b128, back-to-back) ----
  #pragma unroll
  for (int i = 0; i < 4; ++i) {
    int idx = tid + i * 256;                 // 0..1023
    int r = idx >> 3, c4 = (idx & 7) * 4;
    ra[i] = *(const float4*)(A + (size_t)(m0 + r) * lda + c4);
  }
  #pragma unroll
  for (int i = 0; i < 4; ++i) {
    int idx = tid + i * 256;
    int n = idx >> 3, c4 = (idx & 7) * 4;
    rb[i] = *(const float4*)(W + (size_t)(n0 + n) * Kdim + c4);
  }

  for (int k0 = 0; k0 < Kdim; k0 += BK) {
    // ---- pack staged registers -> LDS (bf16, b64 stores) ----
    #pragma unroll
    for (int i = 0; i < 4; ++i) {
      int idx = tid + i * 256;
      int r = idx >> 3, c4 = (idx & 7) * 4;
      u32x2 pk;
      pk[0] = (unsigned)f2bf(ra[i].x) | ((unsigned)f2bf(ra[i].y) << 16);
      pk[1] = (unsigned)f2bf(ra[i].z) | ((unsigned)f2bf(ra[i].w) << 16);
      *(u32x2*)(&As[r][c4]) = pk;
    }
    #pragma unroll
    for (int i = 0; i < 4; ++i) {
      int idx = tid + i * 256;
      int n = idx >> 3, c4 = (idx & 7) * 4;
      u32x2 pk;
      pk[0] = (unsigned)f2bf(rb[i].x) | ((unsigned)f2bf(rb[i].y) << 16);
      pk[1] = (unsigned)f2bf(rb[i].z) | ((unsigned)f2bf(rb[i].w) << 16);
      *(u32x2*)(&Bs[n][c4]) = pk;
    }
    __syncthreads();

    // ---- issue next tile's global loads; consumed next iteration (overlap with WMMA) ----
    const int k1 = k0 + BK;
    if (k1 < Kdim) {
      #pragma unroll
      for (int i = 0; i < 4; ++i) {
        int idx = tid + i * 256;
        int r = idx >> 3, c4 = (idx & 7) * 4;
        ra[i] = *(const float4*)(A + (size_t)(m0 + r) * lda + k1 + c4);
      }
      #pragma unroll
      for (int i = 0; i < 4; ++i) {
        int idx = tid + i * 256;
        int n = idx >> 3, c4 = (idx & 7) * 4;
        rb[i] = *(const float4*)(W + (size_t)(n0 + n) * Kdim + k1 + c4);
      }
      // warm L2 two tiles ahead (global_prefetch_b8; speculative)
      if (k1 + BK < Kdim) {
        __builtin_prefetch(&A[(size_t)(m0 + ldr) * lda + (k1 + BK + ldc4)], 0, 1);
        __builtin_prefetch(&W[(size_t)(n0 + ldr) * Kdim + (k1 + BK + ldc4)], 0, 1);
      }
    }

    // ---- fragments per documented 16-bit wave32 layouts ----
    // A 16x32: lanes 0-15 row M=l15 hold K 0-7 & 16-23; lanes 16-31 hold K 8-15 & 24-31
    FragU a[2];
    #pragma unroll
    for (int i2 = 0; i2 < 2; ++i2) {
      const unsigned short* pa = &As[(wm * 2 + i2) * 16 + l15][0];
      a[i2].q[0] = *(const u32x4*)(pa + hh * 8);
      a[i2].q[1] = *(const u32x4*)(pa + 16 + hh * 8);
    }
    #pragma unroll
    for (int j = 0; j < 4; ++j) {
      // B 32x16: lanes 0-15 col N=l15 hold K 0-7 & 16-23; lanes 16-31 hold K 8-15 & 24-31
      FragU b;
      const unsigned short* pb = &Bs[wn * 64 + j * 16 + l15][0];
      b.q[0] = *(const u32x4*)(pb + hh * 8);
      b.q[1] = *(const u32x4*)(pb + 16 + hh * 8);
      #pragma unroll
      for (int i2 = 0; i2 < 2; ++i2) {
        acc[i2][j] = __builtin_amdgcn_wmma_f32_16x16x32_bf16(
            false, a[i2].v, false, b.v, (short)0, acc[i2][j], false, false);
      }
    }
    __syncthreads();
  }

  // ---- epilogue: C/D layout lane<16 -> M=r, lane>=16 -> M=r+8; N = lane&15 ----
  #pragma unroll
  for (int j = 0; j < 4; ++j) {
    int gn = n0 + wn * 64 + j * 16 + l15;
    float bv = bias ? bias[gn] : 0.0f;
    #pragma unroll
    for (int i2 = 0; i2 < 2; ++i2) {
      #pragma unroll
      for (int r = 0; r < 8; ++r) {
        int gm = m0 + (wm * 2 + i2) * 16 + r + hh * 8;
        size_t off = (size_t)gm * ldc + gn;
        float v = acc[i2][j][r] + bv;
        if (res) v += res[off];
        if (act == 1) v = gelu_f(v);
        C[off] = v;
      }
    }
  }
}

// ---------------- tiny ragged GEMM (final NF=21 head): C[M,N] = A[M,K]*W^T + bias ----------------
__global__ void gemm_small_kernel(const float* __restrict__ A,
                                  const float* __restrict__ W,    // (Ndim, Kdim)
                                  const float* __restrict__ bias,
                                  float* __restrict__ C,
                                  int Mdim, int Ndim, int Kdim)
{
  int idx = blockIdx.x * blockDim.x + threadIdx.x;
  if (idx >= Mdim * Ndim) return;
  int n = idx % Ndim, m = idx / Ndim;
  const float* a = A + (size_t)m * Kdim;
  const float* w = W + (size_t)n * Kdim;
  float s = bias ? bias[n] : 0.0f;
  for (int k = 0; k < Kdim; ++k) s += a[k] * w[k];
  C[(size_t)m * Ndim + n] = s;
}

// ---------------- DFT basis: cos/sin (Mn x Lseq) ----------------
__global__ void dft_basis_kernel(float* __restrict__ cosT, float* __restrict__ sinT)
{
  int idx = blockIdx.x * blockDim.x + threadIdx.x;
  if (idx >= Mn * Lseq) return;
  int t = idx % Lseq, m = idx / Lseq;
  double a = 6.283185307179586476925286766559 * (double)m * (double)t / (double)Lseq;
  cosT[idx] = (float)cos(a);
  sinT[idx] = (float)sin(a);
}

// ---------------- circular conv3 embedding (NF->D) + positional embedding ----------------
__global__ void embed_kernel(const float* __restrict__ X,   // (B,L,NFt)
                             const float* __restrict__ W,   // (Dmod, NFt, 3)
                             float* __restrict__ out, int L)
{
  int idx = blockIdx.x * blockDim.x + threadIdx.x;
  if (idx >= Bsz * L * Dmod) return;
  int d = idx & (Dmod - 1);
  int t = idx >> 9;          // Dmod == 512
  int l = t % L, b = t / L;
  int lm = (l == 0) ? L - 1 : l - 1;
  int lp = (l == L - 1) ? 0 : l + 1;
  const float* x0 = X + ((size_t)b * L + lm) * NFt;
  const float* x1 = X + ((size_t)b * L + l) * NFt;
  const float* x2 = X + ((size_t)b * L + lp) * NFt;
  const float* w  = W + (size_t)d * NFt * 3;
  float s = 0.0f;
  for (int c = 0; c < NFt; ++c)
    s += x0[c] * w[c * 3 + 0] + x1[c] * w[c * 3 + 1] + x2[c] * w[c * 3 + 2];
  float i2 = (float)(d & ~1);
  float f  = expf(-logf(10000.0f) * i2 / (float)Dmod);
  float ang = (float)l * f;
  s += (d & 1) ? cosf(ang) : sinf(ang);
  out[idx] = s;
}

// ---------------- (B,L,H*E) -> (B,H,E,L) transpose ----------------
__global__ void transpose_kernel(const float* __restrict__ X, float* __restrict__ out, int L)
{
  int idx = blockIdx.x * blockDim.x + threadIdx.x;    // out index (b,h,e,l)
  if (idx >= RBHE * Lseq) return;
  int l = idx % L;
  int r = idx / L;
  int e = r & 63, h = (r >> 6) & 7, b = r >> 9;
  out[idx] = X[((size_t)b * L + l) * Dmod + h * En + e];
}

// ---------------- first-64-bin rfft: one wave per (row, mode) ----------------
__global__ void rfft_kernel(const float* __restrict__ X, int L,
                            const float* __restrict__ cosT, const float* __restrict__ sinT,
                            float* __restrict__ outRe, float* __restrict__ outIm)
{
  int gw   = (blockIdx.x * blockDim.x + threadIdx.x) >> 5;
  int lane = threadIdx.x & 31;
  if (gw >= RBHE * Mn) return;
  int r = gw / Mn, m = gw % Mn;
  const float* x  = X + (size_t)r * L;
  const float* cb = cosT + (size_t)m * L;
  const float* sb = sinT + (size_t)m * L;
  float re = 0.0f, im = 0.0f;
  for (int t = lane; t < L; t += 32) {
    float v = x[t];
    re += v * cb[t];
    im -= v * sb[t];
  }
  for (int off = 16; off > 0; off >>= 1) {
    re += __shfl_down(re, off, 32);
    im += __shfl_down(im, off, 32);
  }
  if (lane == 0) { outRe[gw] = re; outIm[gw] = im; }
}

// ---------------- complex per-head mode mixing: of[b,h,o,m] = sum_i x[b,h,i,m]*W[h,i,o,m] ----------------
__global__ void cmodes_kernel(const float* __restrict__ xR, const float* __restrict__ xI,
                              const float* __restrict__ Wf,   // (H,E,E,M,2)
                              float* __restrict__ oR, float* __restrict__ oI)
{
  int idx = blockIdx.x * blockDim.x + threadIdx.x;   // (b,h,o,m)
  if (idx >= RBHE * Mn) return;
  int m = idx & 63, o = (idx >> 6) & 63, h = (idx >> 12) & 7, b = idx >> 15;
  size_t xb = ((size_t)b * Hn + h) * En * Mn;
  float ar = 0.0f, ai = 0.0f;
  for (int i = 0; i < En; ++i) {
    float xr = xR[xb + (size_t)i * Mn + m];
    float xi = xI[xb + (size_t)i * Mn + m];
    const float* w = Wf + ((((size_t)h * En + i) * En + o) * Mn + m) * 2;
    float wr = w[0], wi = w[1];
    ar += xr * wr - xi * wi;
    ai += xr * wi + xi * wr;
  }
  oR[idx] = ar; oI[idx] = ai;
}

// ---------------- irfft from 64 modes ----------------
__global__ __launch_bounds__(256) void irfft_kernel(
    const float* __restrict__ Re, const float* __restrict__ Im,
    const float* __restrict__ cosT, const float* __restrict__ sinT,
    float* __restrict__ out, int L, float scale)
{
  __shared__ float sre[Mn], sim[Mn];
  int row = blockIdx.y;
  if (threadIdx.x < Mn) {
    sre[threadIdx.x] = Re[(size_t)row * Mn + threadIdx.x];
    sim[threadIdx.x] = Im[(size_t)row * Mn + threadIdx.x];
  }
  __syncthreads();
  int t = blockIdx.x * blockDim.x + threadIdx.x;
  if (t >= L) return;
  float s = sre[0];
  for (int m = 1; m < Mn; ++m)
    s += 2.0f * (sre[m] * cosT[(size_t)m * L + t] - sim[m] * sinT[(size_t)m * L + t]);
  out[(size_t)row * L + t] = s * scale;
}

// ---------------- cross attention: s = tanh_c(qf . kf) ----------------
__global__ void cross_s_kernel(const float* __restrict__ qR, const float* __restrict__ qI,
                               const float* __restrict__ kR, const float* __restrict__ kI,
                               float* __restrict__ sR, float* __restrict__ sI)
{
  int idx = blockIdx.x * blockDim.x + threadIdx.x;   // (b,h,x,y)
  if (idx >= Bsz * Hn * Mn * Mn) return;
  int y = idx & 63, x = (idx >> 6) & 63, h = (idx >> 12) & 7, b = idx >> 15;
  size_t base = ((size_t)b * Hn + h) * En * Mn;
  float ar = 0.0f, ai = 0.0f;
  for (int e = 0; e < En; ++e) {
    size_t qo = base + (size_t)e * Mn;
    float q_r = qR[qo + x], q_i = qI[qo + x];
    float k_r = kR[qo + y], k_i = kI[qo + y];
    ar += q_r * k_r - q_i * k_i;
    ai += q_r * k_i + q_i * k_r;
  }
  // complex tanh(a+bi) = (sinh 2a + i sin 2b) / (cosh 2a + cos 2b)
  float den = coshf(2.0f * ar) + cosf(2.0f * ai);
  sR[idx] = sinhf(2.0f * ar) / den;
  sI[idx] = sinf(2.0f * ai) / den;
}

// o[b,h,e,x] = sum_y s[b,h,x,y] * kf[b,h,e,y]
__global__ void cross_o_kernel(const float* __restrict__ sR, const float* __restrict__ sI,
                               const float* __restrict__ kR, const float* __restrict__ kI,
                               float* __restrict__ oR, float* __restrict__ oI)
{
  int idx = blockIdx.x * blockDim.x + threadIdx.x;   // (b,h,e,x)
  if (idx >= RBHE * Mn) return;
  int x = idx & 63, e = (idx >> 6) & 63, h = (idx >> 12) & 7, b = idx >> 15;
  size_t sb = (((size_t)b * Hn + h) * Mn + x) * Mn;
  size_t kb = (((size_t)b * Hn + h) * En + e) * Mn;
  float ar = 0.0f, ai = 0.0f;
  for (int y = 0; y < Mn; ++y) {
    float s_r = sR[sb + y], s_i = sI[sb + y];
    float k_r = kR[kb + y], k_i = kI[kb + y];
    ar += s_r * k_r - s_i * k_i;
    ai += s_r * k_i + s_i * k_r;
  }
  oR[idx] = ar; oI[idx] = ai;
}

// ---------------- series_decomp: seasonal = x - ma25(x); trend modes: 0 skip, 1 write, 2 accumulate ----------------
__global__ void decomp_kernel(const float* __restrict__ X, float* __restrict__ seas,
                              float* __restrict__ trend, int L, int Cc, int mode)
{
  int idx = blockIdx.x * blockDim.x + threadIdx.x;
  if (idx >= Bsz * L * Cc) return;
  int c = idx % Cc;
  int t = idx / Cc;
  int l = t % L, b = t / L;
  const float* xb = X + (size_t)b * L * Cc + c;
  float s = 0.0f;
  for (int j = 0; j < MAw; ++j) {
    int p = l + j - (MAw - 1) / 2;
    p = p < 0 ? 0 : (p >= L ? L - 1 : p);
    s += xb[(size_t)p * Cc];
  }
  s *= (1.0f / (float)MAw);
  seas[idx] = X[idx] - s;
  if (mode == 1) trend[idx] = s;
  else if (mode == 2) trend[idx] += s;
}

// ---------------- column mean over time (B,L,C) -> (B,C) ----------------
__global__ __launch_bounds__(256) void colmean_kernel(const float* __restrict__ X,
                                                      float* __restrict__ out, int L, int Cc)
{
  __shared__ float red[256];
  int b = blockIdx.x / Cc, c = blockIdx.x % Cc;
  float s = 0.0f;
  for (int l = threadIdx.x; l < L; l += 256) s += X[((size_t)b * L + l) * Cc + c];
  red[threadIdx.x] = s;
  __syncthreads();
  for (int st = 128; st > 0; st >>= 1) {
    if (threadIdx.x < st) red[threadIdx.x] += red[threadIdx.x + st];
    __syncthreads();
  }
  if (threadIdx.x == 0) out[blockIdx.x] = red[0] / (float)L;
}

// ---------------- decoder seasonal/trend init ----------------
__global__ void dec_init_kernel(const float* __restrict__ Se, const float* __restrict__ Tr,
                                const float* __restrict__ meanBC,
                                float* __restrict__ SeI, float* __restrict__ TrI)
{
  int idx = blockIdx.x * blockDim.x + threadIdx.x;
  if (idx >= Bsz * Ldec * NFt) return;
  int c = idx % NFt;
  int t = idx / NFt;
  int l = t % Ldec, b = t / Ldec;
  if (l < Llab) {
    size_t src = ((size_t)b * Lseq + (Lseq - Llab + l)) * NFt + c;
    SeI[idx] = Se[src];
    TrI[idx] = Tr[src];
  } else {
    SeI[idx] = 0.0f;
    TrI[idx] = meanBC[b * NFt + c];
  }
}

// ---------------- layernorm over D per row ----------------
__global__ __launch_bounds__(256) void layernorm_kernel(const float* __restrict__ X,
    const float* __restrict__ g, const float* __restrict__ be, float* __restrict__ out)
{
  __shared__ float r1[256], r2[256];
  size_t row = blockIdx.x;
  const float* x = X + row * Dmod;
  float s1 = 0.0f, s2 = 0.0f;
  for (int d = threadIdx.x; d < Dmod; d += 256) { float v = x[d]; s1 += v; s2 += v * v; }
  r1[threadIdx.x] = s1; r2[threadIdx.x] = s2;
  __syncthreads();
  for (int st = 128; st > 0; st >>= 1) {
    if (threadIdx.x < st) { r1[threadIdx.x] += r1[threadIdx.x + st]; r2[threadIdx.x] += r2[threadIdx.x + st]; }
    __syncthreads();
  }
  float mu  = r1[0] / (float)Dmod;
  float var = r2[0] / (float)Dmod - mu * mu;
  float inv = rsqrtf(var + 1e-5f);
  for (int d = threadIdx.x; d < Dmod; d += 256)
    out[row * Dmod + d] = (x[d] - mu) * inv * g[d] + be[d];
}

// ---------------- my_Layernorm part 2: subtract per-(b,d) mean over time ----------------
__global__ __launch_bounds__(256) void timemean_sub_kernel(const float* __restrict__ X,
                                                           float* __restrict__ out, int L)
{
  __shared__ float red[256];
  int b = blockIdx.x / Dmod, d = blockIdx.x % Dmod;
  const float* x = X + (size_t)b * L * Dmod + d;
  float s = 0.0f;
  for (int l = threadIdx.x; l < L; l += 256) s += x[(size_t)l * Dmod];
  red[threadIdx.x] = s;
  __syncthreads();
  for (int st = 128; st > 0; st >>= 1) {
    if (threadIdx.x < st) red[threadIdx.x] += red[threadIdx.x + st];
    __syncthreads();
  }
  float mu = red[0] / (float)L;
  float* o = out + (size_t)b * L * Dmod + d;
  for (int l = threadIdx.x; l < L; l += 256) o[(size_t)l * Dmod] = x[(size_t)l * Dmod] - mu;
}

// ---------------- trend projection: circular conv3 D->NF ----------------
__global__ void conv3_trend_kernel(const float* __restrict__ X,   // (B,Ldec,Dmod)
                                   const float* __restrict__ W,   // (NFt, Dmod, 3)
                                   float* __restrict__ out)
{
  int idx = blockIdx.x * blockDim.x + threadIdx.x;
  if (idx >= Bsz * Ldec * NFt) return;
  int c = idx % NFt;
  int t = idx / NFt;
  int l = t % Ldec, b = t / Ldec;
  int lm = (l == 0) ? Ldec - 1 : l - 1;
  int lp = (l == Ldec - 1) ? 0 : l + 1;
  const float* x0 = X + ((size_t)b * Ldec + lm) * Dmod;
  const float* x1 = X + ((size_t)b * Ldec + l) * Dmod;
  const float* x2 = X + ((size_t)b * Ldec + lp) * Dmod;
  const float* w  = W + (size_t)c * Dmod * 3;
  float s = 0.0f;
  for (int d = 0; d < Dmod; ++d)
    s += x0[d] * w[d * 3 + 0] + x1[d] * w[d * 3 + 1] + x2[d] * w[d * 3 + 2];
  out[idx] = s;
}

__global__ void zero_kernel(float* __restrict__ p, size_t n)
{
  size_t i = (size_t)blockIdx.x * blockDim.x + threadIdx.x;
  if (i < n) p[i] = 0.0f;
}

__global__ void final_kernel(const float* __restrict__ TrI, const float* __restrict__ Rt,
                             const float* __restrict__ Ssea, float* __restrict__ out)
{
  int idx = blockIdx.x * blockDim.x + threadIdx.x;
  if (idx >= Bsz * Lpred * NFt) return;
  int c = idx % NFt;
  int t = idx / NFt;
  int p = t % Lpred, b = t / Lpred;
  size_t src = ((size_t)b * Ldec + (Llab + p)) * NFt + c;
  out[idx] = TrI[src] + Rt[src] + Ssea[src];
}

// ---------------- host-side GEMM launcher (full-tile WMMA path) ----------------
static inline void gemm(hipStream_t s, const float* A, const float* W, const float* bias,
                        const float* res, float* C, int Mdim, int Ndim, int Kdim, int act)
{
  dim3 g(Ndim / BN, Mdim / BM);
  gemm_wmma_kernel<<<g, 256, 0, s>>>(A, Kdim, W, bias, res, C, Ndim, Kdim, act);
}

extern "C" void kernel_launch(void* const* d_in, const int* in_sizes, int n_in,
                              void* d_out, int out_size, void* d_ws, size_t ws_size,
                              hipStream_t stream)
{
  (void)in_sizes; (void)n_in; (void)out_size; (void)ws_size;
  const float* x_enc      = (const float*)d_in[0];
  const float* enc_emb_W  = (const float*)d_in[3];
  const float* dec_emb_W  = (const float*)d_in[4];
  const float* enc_qkvo_W = (const float*)d_in[5];
  const float* enc_qkvo_b = (const float*)d_in[6];
  const float* enc_ffn_W1 = (const float*)d_in[7];
  const float* enc_ffn_W2 = (const float*)d_in[8];
  const float* enc_fourier_W = (const float*)d_in[9];
  const float* enc_norm_g = (const float*)d_in[10];
  const float* enc_norm_b = (const float*)d_in[11];
  const float* dec_qkvo_W = (const float*)d_in[12];
  const float* dec_qkvo_b = (const float*)d_in[13];
  const float* dec_ffn_W1 = (const float*)d_in[14];
  const float* dec_ffn_W2 = (const float*)d_in[15];
  const float* dec_self_fourier_W  = (const float*)d_in[16];
  const float* dec_cross_fourier_W = (const float*)d_in[17];
  const float* dec_trend_proj_W    = (const float*)d_in[18];
  const float* dec_norm_g = (const float*)d_in[19];
  const float* dec_norm_b = (const float*)d_in[20];
  const float* final_W    = (const float*)d_in[21];
  const float* final_b    = (const float*)d_in[22];
  float* out = (float*)d_out;

  // ---- workspace arena ----
  float* ws = (float*)d_ws;
  size_t off = 0;
  auto alloc = [&](size_t n) { float* p = ws + off; off += n; return p; };
  const size_t N0 = (size_t)BLEN * Dmod;
  float* X0 = alloc(N0);                       // enc stream / enc_out
  float* X1 = alloc(N0);                       // temp (also decoder temp)
  float* X2 = alloc(N0);                       // temp (B,H,E,L) == (B*L,D) view
  float* Y0 = alloc(N0);                       // decoder stream
  float* TA = alloc(N0);                       // accumulated decoder trend
  float* FM = alloc((size_t)BLEN * Dff);       // FFN mid
  float* cosT = alloc((size_t)Mn * Lseq);
  float* sinT = alloc((size_t)Mn * Lseq);
  const size_t RM = (size_t)RBHE * Mn;
  float* qfR = alloc(RM); float* qfI = alloc(RM);
  float* kfR = alloc(RM); float* kfI = alloc(RM);
  float* ofR = alloc(RM); float* ofI = alloc(RM);
  float* oxR = alloc(RM); float* oxI = alloc(RM);
  const size_t SMM = (size_t)Bsz * Hn * Mn * Mn;
  float* sR = alloc(SMM); float* sI = alloc(SMM);
  const size_t SNF = (size_t)Bsz * Ldec * NFt;
  float* Se  = alloc(SNF); float* Tr  = alloc(SNF);
  float* SeI = alloc(SNF); float* TrI = alloc(SNF);
  float* Rt  = alloc(SNF); float* Ssea = alloc(SNF);
  float* meanBC = alloc((size_t)Bsz * NFt);

  // ---- DFT basis (SEQ == LDEC == 4096, one table serves both) ----
  dft_basis_kernel<<<CEILDIV(Mn * Lseq, 256), 256, 0, stream>>>(cosT, sinT);

  // ---- encoder embedding ----
  embed_kernel<<<CEILDIV(Bsz * Lseq * Dmod, 256), 256, 0, stream>>>(x_enc, enc_emb_W, X0, Lseq);

  // ---- encoder layers (k,v projections unused by Fourier attention -> skipped) ----
  for (int l = 0; l < 2; ++l) {
    const float* Wq = enc_qkvo_W + (size_t)(l * 4 + 0) * Dmod * Dmod;
    const float* bq = enc_qkvo_b + (l * 4 + 0) * Dmod;
    const float* Wo = enc_qkvo_W + (size_t)(l * 4 + 3) * Dmod * Dmod;
    const float* bo = enc_qkvo_b + (l * 4 + 3) * Dmod;
    gemm(stream, X0, Wq, bq, nullptr, X1, BLEN, Dmod, Dmod, 0);
    transpose_kernel<<<CEILDIV(RBHE * Lseq, 256), 256, 0, stream>>>(X1, X2, Lseq);
    rfft_kernel<<<CEILDIV(RBHE * Mn * 32, 256), 256, 0, stream>>>(X2, Lseq, cosT, sinT, qfR, qfI);
    cmodes_kernel<<<CEILDIV((int)RM, 256), 256, 0, stream>>>(qfR, qfI, enc_fourier_W, ofR, ofI);
    { dim3 g(CEILDIV(Lseq, 256), RBHE);
      irfft_kernel<<<g, 256, 0, stream>>>(ofR, ofI, cosT, sinT, X2, Lseq, 1.0f / (float)Lseq); }
    gemm(stream, X2, Wo, bo, /*res*/X0, X1, BLEN, Dmod, Dmod, 0);          // x + attn
    decomp_kernel<<<CEILDIV(Bsz * Lseq * Dmod, 256), 256, 0, stream>>>(X1, X0, nullptr, Lseq, Dmod, 0);
    gemm(stream, X0, enc_ffn_W1 + (size_t)l * Dff * Dmod, nullptr, nullptr, FM, BLEN, Dff, Dmod, 1);
    gemm(stream, FM, enc_ffn_W2 + (size_t)l * Dmod * Dff, nullptr, /*res*/X0, X1, BLEN, Dmod, Dff, 0);
    decomp_kernel<<<CEILDIV(Bsz * Lseq * Dmod, 256), 256, 0, stream>>>(X1, X0, nullptr, Lseq, Dmod, 0);
  }
  // my_layernorm(enc_out)
  layernorm_kernel<<<BLEN, 256, 0, stream>>>(X0, enc_norm_g, enc_norm_b, X1);
  timemean_sub_kernel<<<Bsz * Dmod, 256, 0, stream>>>(X1, X0, Lseq);

  // ---- decoder init ----
  decomp_kernel<<<CEILDIV(Bsz * Lseq * NFt, 256), 256, 0, stream>>>(x_enc, Se, Tr, Lseq, NFt, 1);
  colmean_kernel<<<Bsz * NFt, 256, 0, stream>>>(x_enc, meanBC, Lseq, NFt);
  dec_init_kernel<<<CEILDIV(Bsz * Ldec * NFt, 256), 256, 0, stream>>>(Se, Tr, meanBC, SeI, TrI);
  embed_kernel<<<CEILDIV(Bsz * Ldec * Dmod, 256), 256, 0, stream>>>(SeI, dec_emb_W, Y0, Ldec);
  zero_kernel<<<CEILDIV((int)N0, 256), 256, 0, stream>>>(TA, N0);

  // ---- decoder layer ----
  {
    // self attention (Fourier)
    const float* Wq = dec_qkvo_W + (size_t)(0 * 4 + 0) * Dmod * Dmod;
    const float* bq = dec_qkvo_b + (0 * 4 + 0) * Dmod;
    const float* Wo = dec_qkvo_W + (size_t)(0 * 4 + 3) * Dmod * Dmod;
    const float* bo = dec_qkvo_b + (0 * 4 + 3) * Dmod;
    gemm(stream, Y0, Wq, bq, nullptr, X1, BLEN, Dmod, Dmod, 0);
    transpose_kernel<<<CEILDIV(RBHE * Ldec, 256), 256, 0, stream>>>(X1, X2, Ldec);
    rfft_kernel<<<CEILDIV(RBHE * Mn * 32, 256), 256, 0, stream>>>(X2, Ldec, cosT, sinT, qfR, qfI);
    cmodes_kernel<<<CEILDIV((int)RM, 256), 256, 0, stream>>>(qfR, qfI, dec_self_fourier_W, ofR, ofI);
    { dim3 g(CEILDIV(Ldec, 256), RBHE);
      irfft_kernel<<<g, 256, 0, stream>>>(ofR, ofI, cosT, sinT, X2, Ldec, 1.0f / (float)Ldec); }
    gemm(stream, X2, Wo, bo, /*res*/Y0, X1, BLEN, Dmod, Dmod, 0);
    decomp_kernel<<<CEILDIV(Bsz * Ldec * Dmod, 256), 256, 0, stream>>>(X1, Y0, TA, Ldec, Dmod, 2);

    // cross attention (Fourier)
    const float* Wcq = dec_qkvo_W + (size_t)(1 * 4 + 0) * Dmod * Dmod;
    const float* bcq = dec_qkvo_b + (1 * 4 + 0) * Dmod;
    const float* Wck = dec_qkvo_W + (size_t)(1 * 4 + 1) * Dmod * Dmod;
    const float* bck = dec_qkvo_b + (1 * 4 + 1) * Dmod;
    const float* Wco = dec_qkvo_W + (size_t)(1 * 4 + 3) * Dmod * Dmod;
    const float* bco = dec_qkvo_b + (1 * 4 + 3) * Dmod;
    gemm(stream, Y0, Wcq, bcq, nullptr, X1, BLEN, Dmod, Dmod, 0);
    transpose_kernel<<<CEILDIV(RBHE * Ldec, 256), 256, 0, stream>>>(X1, X2, Ldec);
    rfft_kernel<<<CEILDIV(RBHE * Mn * 32, 256), 256, 0, stream>>>(X2, Ldec, cosT, sinT, qfR, qfI);
    gemm(stream, X0, Wck, bck, nullptr, X1, BLEN, Dmod, Dmod, 0);        // k from enc_out
    transpose_kernel<<<CEILDIV(RBHE * Lseq, 256), 256, 0, stream>>>(X1, X2, Lseq);
    rfft_kernel<<<CEILDIV(RBHE * Mn * 32, 256), 256, 0, stream>>>(X2, Lseq, cosT, sinT, kfR, kfI);
    cross_s_kernel<<<CEILDIV((int)SMM, 256), 256, 0, stream>>>(qfR, qfI, kfR, kfI, sR, sI);
    cross_o_kernel<<<CEILDIV((int)RM, 256), 256, 0, stream>>>(sR, sI, kfR, kfI, oxR, oxI);
    cmodes_kernel<<<CEILDIV((int)RM, 256), 256, 0, stream>>>(oxR, oxI, dec_cross_fourier_W, ofR, ofI);
    { dim3 g(CEILDIV(Ldec, 256), RBHE);
      float sc = (1.0f / (float)Ldec) / ((float)Dmod * (float)Dmod);     // irfft * 1/(in*out)
      irfft_kernel<<<g, 256, 0, stream>>>(ofR, ofI, cosT, sinT, X2, Ldec, sc); }
    gemm(stream, X2, Wco, bco, /*res*/Y0, X1, BLEN, Dmod, Dmod, 0);
    decomp_kernel<<<CEILDIV(Bsz * Ldec * Dmod, 256), 256, 0, stream>>>(X1, Y0, TA, Ldec, Dmod, 2);

    // FFN
    gemm(stream, Y0, dec_ffn_W1, nullptr, nullptr, FM, BLEN, Dff, Dmod, 1);
    gemm(stream, FM, dec_ffn_W2, nullptr, /*res*/Y0, X1, BLEN, Dmod, Dff, 0);
    decomp_kernel<<<CEILDIV(Bsz * Ldec * Dmod, 256), 256, 0, stream>>>(X1, Y0, TA, Ldec, Dmod, 2);
  }

  // residual trend projection (t1+t2+t3 accumulated in TA)
  conv3_trend_kernel<<<CEILDIV(Bsz * Ldec * NFt, 256), 256, 0, stream>>>(TA, dec_trend_proj_W, Rt);

  // seasonal head: my_layernorm(x) @ final_W^T + final_b  (ragged N=21 -> scalar kernel)
  layernorm_kernel<<<BLEN, 256, 0, stream>>>(Y0, dec_norm_g, dec_norm_b, X1);
  timemean_sub_kernel<<<Bsz * Dmod, 256, 0, stream>>>(X1, X2, Ldec);
  gemm_small_kernel<<<CEILDIV(BLEN * NFt, 256), 256, 0, stream>>>(X2, final_W, final_b, Ssea,
                                                                  BLEN, NFt, Dmod);

  // out = (trend_init + residual_trend + seasonal)[:, -PRED:, :]
  final_kernel<<<CEILDIV(Bsz * Lpred * NFt, 256), 256, 0, stream>>>(TrI, Rt, Ssea, out);
}
